// ShowAttendTellModel_4604204941771
// MI455X (gfx1250) — matrix operand order repaired and checked by reading, not compiled, not run on hardware
//
#include <hip/hip_runtime.h>
#include <hip/hip_bf16.h>
#include <stdint.h>

typedef __attribute__((ext_vector_type(16))) __bf16 v16bf;
typedef __attribute__((ext_vector_type(8)))  float  v8f;

namespace {
constexpr int kB  = 128;        // batch
constexpr int kT  = 17;         // seq columns (loop runs kT-1)
constexpr int kV1 = 9488;       // vocab + 1
constexpr int kE  = 512;        // embed dim
constexpr int kH  = 512;        // rnn size
constexpr int kF  = 2048;       // feat size
constexpr int kA  = 196;        // att hid
constexpr int kL  = 196;        // att spatial
constexpr int kBL = kB * kL;    // 25088 rows of p_att
constexpr int k3H = 3 * kH;     // 1536
constexpr int kEF = kE + kF;    // 2560

__device__ __forceinline__ __bf16 f2bf(float f) {
  union { float f; uint32_t u; } x; x.f = f;
  uint32_t r = x.u + 0x7fffu + ((x.u >> 16) & 1u);   // round-to-nearest-even
  union { unsigned short s; __bf16 b; } o; o.s = (unsigned short)(r >> 16);
  return o.b;
}
__device__ __forceinline__ float bf2f(__bf16 b) {
  union { unsigned short s; __bf16 b; } i; i.b = b;
  union { uint32_t u; float f; } o; o.u = ((uint32_t)i.s) << 16;
  return o.f;
}
__device__ __forceinline__ float sigmoidf_(float x) {
  return 1.0f / (1.0f + __expf(-x));
}
union BF16Tile { v16bf v; uint4 q[2]; };
} // namespace

// ---------------------------------------------------------------------------
// Generic bf16 WMMA GEMM: C[M,N] = A[M,K] * W[N,K]^T + bias[N]
// A, W row-major bf16. One wave per block; wave computes a 16 x (16*NT) strip.
// Branch-free K loop: out-of-range B columns are CLAMPED to N-1 (a B lane only
// feeds its own output column, and columns >= N are never stored), so no
// zero-fill or EXEC juggling is needed in the hot loop.
// ---------------------------------------------------------------------------
template <int NT>
__global__ __launch_bounds__(32)
void gemm_bf16(const __bf16* __restrict__ A, const __bf16* __restrict__ W,
               const float* __restrict__ bias,
               float* __restrict__ C, __bf16* __restrict__ Cbf,
               int N, int K, int lda, int ldw, long long ldc) {
  const int lane = threadIdx.x & 31;
  const int half = lane >> 4;   // 0: lanes 0-15, 1: lanes 16-31
  const int lid  = lane & 15;
  const int row  = blockIdx.y << 4;
  const int nb   = blockIdx.x * (16 * NT);

  v8f acc[NT] = {};

  // A operand (16x32 bf16): lane<16 -> row m=row+lid, K chunks {0..7, 16..23};
  //                         lane>=16 -> same rows, K chunks {8..15, 24..31}.
  const __bf16* arow = A + (size_t)(row + lid) * lda + (half << 3);

  // B operand (32x16 bf16): lane<16 -> col n, K chunks {0..7, 16..23};
  //                         lane>=16 -> col n, K chunks {8..15, 24..31}.
  const __bf16* bptr[NT];
#pragma unroll
  for (int t = 0; t < NT; ++t) {
    int ncol = nb + t * 16 + lid;
    ncol = ncol < N ? ncol : N - 1;          // clamp; junk cols never stored
    bptr[t] = W + (size_t)ncol * ldw + (half << 3);
  }

  for (int k0 = 0; k0 < K; k0 += 32) {
    BF16Tile a;
    a.q[0] = *(const uint4*)(arow + k0);
    a.q[1] = *(const uint4*)(arow + k0 + 16);
#pragma unroll
    for (int t = 0; t < NT; ++t) {
      BF16Tile bm;
      bm.q[0] = *(const uint4*)(bptr[t] + k0);
      bm.q[1] = *(const uint4*)(bptr[t] + k0 + 16);
      acc[t] = __builtin_amdgcn_wmma_f32_16x16x32_bf16(
          false, a.v, false, bm.v, (short)0, acc[t], false, false);
    }
  }

  // C layout: VGPR r, lanes 0-15 -> (row+r, col), lanes 16-31 -> (row+8+r, col)
  const int rbase = row + (half << 3);
#pragma unroll
  for (int t = 0; t < NT; ++t) {
    const int col = nb + t * 16 + lid;
    if (col < N) {
      const float bv = bias ? bias[col] : 0.0f;
#pragma unroll
      for (int r = 0; r < 8; ++r) {
        const float v = acc[t][r] + bv;
        C[(size_t)(rbase + r) * ldc + col] = v;
        if (Cbf) Cbf[(size_t)(rbase + r) * ldc + col] = f2bf(v);
      }
    }
  }
}

// ---------------------------------------------------------------------------
__global__ void cvt_f32_bf16_kernel(const float* __restrict__ src,
                                    __bf16* __restrict__ dst, long long n) {
  long long i = (long long)blockIdx.x * blockDim.x + threadIdx.x;
  if (i < n) dst[i] = f2bf(src[i]);
}

// dot[b,l] = sum_a tanh(p_att[b,l,a] + att_h[b,a]) * alpha_W[a] + alpha_b
// w[b,l]   = softmax_l(dot)
__global__ void attn_softmax_kernel(const float* __restrict__ p_att,
                                    const float* __restrict__ att_h,
                                    const float* __restrict__ alpha_W,
                                    const float* __restrict__ alpha_b,
                                    float* __restrict__ w_out) {
  __shared__ float sh_ah[kA];
  __shared__ float sh_aw[kA];
  __shared__ float sdot[kL];
  __shared__ float red[256];
  const int b = blockIdx.x, tid = threadIdx.x;
  for (int i = tid; i < kA; i += 256) {
    sh_ah[i] = att_h[b * kA + i];
    sh_aw[i] = alpha_W[i];
  }
  __syncthreads();
  const float ab = alpha_b[0];
  for (int l = tid; l < kL; l += 256) {
    const float* p = p_att + ((size_t)b * kL + l) * kA;
    float s = ab;
    for (int a = 0; a < kA; ++a) s = fmaf(tanhf(p[a] + sh_ah[a]), sh_aw[a], s);
    sdot[l] = s;
  }
  __syncthreads();
  float m = -3.4e38f;
  for (int l = tid; l < kL; l += 256) m = fmaxf(m, sdot[l]);
  red[tid] = m;
  __syncthreads();
  for (int s = 128; s > 0; s >>= 1) {
    if (tid < s) red[tid] = fmaxf(red[tid], red[tid + s]);
    __syncthreads();
  }
  m = red[0];
  __syncthreads();
  float sum = 0.0f;
  for (int l = tid; l < kL; l += 256) {
    const float e = __expf(sdot[l] - m);
    sdot[l] = e;
    sum += e;
  }
  red[tid] = sum;
  __syncthreads();
  for (int s = 128; s > 0; s >>= 1) {
    if (tid < s) red[tid] += red[tid + s];
    __syncthreads();
  }
  const float inv = 1.0f / red[0];
  for (int l = tid; l < kL; l += 256) w_out[b * kL + l] = sdot[l] * inv;
}

// x_bf[b, 0:512] = bf16(embed_W[seq[b,t], :])
__global__ void embed_gather_kernel(const int* __restrict__ seq,
                                    const float* __restrict__ embed_W,
                                    __bf16* __restrict__ x_bf, int t) {
  const int b = blockIdx.y;
  const int e = blockIdx.x * 256 + threadIdx.x;
  const int tok = seq[b * kT + t];
  x_bf[(size_t)b * kEF + e] = f2bf(embed_W[(size_t)tok * kE + e]);
}

// x_bf[b, 512+f] = bf16( sum_l w[b,l] * att_bf[b,l,f] )   (bf16 att stays in L2)
__global__ void att_apply_kernel(const float* __restrict__ w,
                                 const __bf16* __restrict__ att_bf,
                                 __bf16* __restrict__ x_bf) {
  __shared__ float sw[kL];
  const int b = blockIdx.y, tid = threadIdx.x;
  for (int i = tid; i < kL; i += 256) sw[i] = w[b * kL + i];
  __syncthreads();
  const int f = blockIdx.x * 256 + tid;
  const __bf16* base = att_bf + (size_t)b * kL * kF + f;
  float acc = 0.0f;
#pragma unroll 4
  for (int l = 0; l < kL; ++l) acc = fmaf(sw[l], bf2f(base[(size_t)l * kF]), acc);
  x_bf[(size_t)b * kEF + kE + f] = f2bf(acc);
}

// GRU gates: h_new = (1-z)*n + z*h ; writes f32 h and bf16 h
__global__ void gru_gate_kernel(const float* __restrict__ gi,
                                const float* __restrict__ gh,
                                float* __restrict__ h, __bf16* __restrict__ h_bf) {
  const int idx = blockIdx.x * 256 + threadIdx.x;
  const int b = idx >> 9;
  const int j = idx & (kH - 1);
  const float* gib = gi + (size_t)b * k3H;
  const float* ghb = gh + (size_t)b * k3H;
  const float r = sigmoidf_(gib[j] + ghb[j]);
  const float z = sigmoidf_(gib[kH + j] + ghb[kH + j]);
  const float n = tanhf(gib[2 * kH + j] + r * ghb[2 * kH + j]);
  const float ho = h[(size_t)b * kH + j];
  const float hn = (1.0f - z) * n + z * ho;
  h[(size_t)b * kH + j] = hn;
  h_bf[(size_t)b * kH + j] = f2bf(hn);
}

// in-place log-softmax over out[b, t, :]
__global__ void log_softmax_kernel(float* __restrict__ out, int t) {
  const int b = blockIdx.x, tid = threadIdx.x;
  float* row = out + ((size_t)b * (kT - 1) + t) * kV1;
  __shared__ float red[256];
  float m = -3.4e38f;
  for (int v = tid; v < kV1; v += 256) m = fmaxf(m, row[v]);
  red[tid] = m;
  __syncthreads();
  for (int s = 128; s > 0; s >>= 1) {
    if (tid < s) red[tid] = fmaxf(red[tid], red[tid + s]);
    __syncthreads();
  }
  m = red[0];
  __syncthreads();
  float sum = 0.0f;
  for (int v = tid; v < kV1; v += 256) sum += __expf(row[v] - m);
  red[tid] = sum;
  __syncthreads();
  for (int s = 128; s > 0; s >>= 1) {
    if (tid < s) red[tid] += red[tid + s];
    __syncthreads();
  }
  const float lse = m + __logf(red[0]);
  for (int v = tid; v < kV1; v += 256) row[v] -= lse;
}

// ---------------------------------------------------------------------------
extern "C" void kernel_launch(void* const* d_in, const int* in_sizes, int n_in,
                              void* d_out, int out_size, void* d_ws, size_t ws_size,
                              hipStream_t stream) {
  const float* fc_feats  = (const float*)d_in[0];
  const float* att_feats = (const float*)d_in[1];
  const int*   seq       = (const int*)  d_in[2];
  const float* embed_W   = (const float*)d_in[3];
  const float* linear_W  = (const float*)d_in[4];
  const float* linear_b  = (const float*)d_in[5];
  const float* ctx2att_W = (const float*)d_in[6];
  const float* ctx2att_b = (const float*)d_in[7];
  const float* h2att_W   = (const float*)d_in[8];
  const float* h2att_b   = (const float*)d_in[9];
  const float* alpha_W   = (const float*)d_in[10];
  const float* alpha_b   = (const float*)d_in[11];
  const float* W_ih      = (const float*)d_in[12];
  const float* W_hh      = (const float*)d_in[13];
  const float* logit_W   = (const float*)d_in[14];
  const float* logit_b   = (const float*)d_in[15];
  float* out = (float*)d_out;
  char* wsb = (char*)d_ws;

  size_t off = 0;
  auto carve = [&](size_t bytes) -> char* {
    char* p = wsb + off;
    off += (bytes + 255) & ~(size_t)255;
    return p;
  };

  __bf16* att_bf = (__bf16*)carve((size_t)kBL * kF * 2);   // 103 MB (L2-resident)
  float*  p_att  = (float*) carve((size_t)kBL * kA * 4);   // 19.7 MB
  __bf16* lin_bf = (__bf16*)carve((size_t)kH  * kF  * 2);
  __bf16* ctx_bf = (__bf16*)carve((size_t)kA  * kF  * 2);
  __bf16* h2a_bf = (__bf16*)carve((size_t)kA  * kH  * 2);
  __bf16* wih_bf = (__bf16*)carve((size_t)k3H * kEF * 2);
  __bf16* whh_bf = (__bf16*)carve((size_t)k3H * kH  * 2);
  __bf16* lgt_bf = (__bf16*)carve((size_t)kV1 * kH  * 2);
  __bf16* fc_bf  = (__bf16*)carve((size_t)kB  * kF  * 2);
  float*  h_f    = (float*) carve((size_t)kB  * kH  * 4);
  __bf16* h_bf   = (__bf16*)carve((size_t)kB  * kH  * 2);
  float*  att_h  = (float*) carve((size_t)kB  * kA  * 4);
  float*  w_soft = (float*) carve((size_t)kB  * kL  * 4);
  __bf16* x_bf   = (__bf16*)carve((size_t)kB  * kEF * 2);
  float*  gi     = (float*) carve((size_t)kB  * k3H * 4);
  float*  gh     = (float*) carve((size_t)kB  * k3H * 4);

  auto cvt = [&](const float* s, __bf16* d, long long n) {
    cvt_f32_bf16_kernel<<<(int)((n + 255) / 256), 256, 0, stream>>>(s, d, n);
  };
  cvt(att_feats, att_bf, (long long)kBL * kF);
  cvt(linear_W,  lin_bf, (long long)kH  * kF);
  cvt(ctx2att_W, ctx_bf, (long long)kA  * kF);
  cvt(h2att_W,   h2a_bf, (long long)kA  * kH);
  cvt(W_ih,      wih_bf, (long long)k3H * kEF);
  cvt(W_hh,      whh_bf, (long long)k3H * kH);
  cvt(logit_W,   lgt_bf, (long long)kV1 * kH);
  cvt(fc_feats,  fc_bf,  (long long)kB  * kF);

  // h0 = fc @ linear_W^T + b  (writes both f32 and bf16 h)
  gemm_bf16<1><<<dim3(kH / 16, kB / 16), 32, 0, stream>>>(
      fc_bf, lin_bf, linear_b, h_f, h_bf, kH, kF, kF, kF, (long long)kH);

  // p_att = att_feats @ ctx2att_W^T + b   (M=25088, N=196, K=2048)
  gemm_bf16<4><<<dim3((kA + 63) / 64, kBL / 16), 32, 0, stream>>>(
      att_bf, ctx_bf, ctx2att_b, p_att, (__bf16*)nullptr, kA, kF, kF, kF, (long long)kA);

  for (int t = 0; t < kT - 1; ++t) {
    // att_h = h @ h2att_W^T + b
    gemm_bf16<1><<<dim3((kA + 15) / 16, kB / 16), 32, 0, stream>>>(
        h_bf, h2a_bf, h2att_b, att_h, (__bf16*)nullptr, kA, kH, kH, kH, (long long)kA);

    attn_softmax_kernel<<<kB, 256, 0, stream>>>(p_att, att_h, alpha_W, alpha_b, w_soft);

    embed_gather_kernel<<<dim3(kE / 256, kB), 256, 0, stream>>>(seq, embed_W, x_bf, t);

    att_apply_kernel<<<dim3(kF / 256, kB), 256, 0, stream>>>(w_soft, att_bf, x_bf);

    // gi = x @ W_ih^T   (K = 2560, concat order [embed, att_res] matches W_ih)
    gemm_bf16<1><<<dim3(k3H / 16, kB / 16), 32, 0, stream>>>(
        x_bf, wih_bf, (const float*)nullptr, gi, (__bf16*)nullptr,
        k3H, kEF, kEF, kEF, (long long)k3H);

    // gh = h @ W_hh^T
    gemm_bf16<1><<<dim3(k3H / 16, kB / 16), 32, 0, stream>>>(
        h_bf, whh_bf, (const float*)nullptr, gh, (__bf16*)nullptr,
        k3H, kH, kH, kH, (long long)k3H);

    gru_gate_kernel<<<(kB * kH) / 256, 256, 0, stream>>>(gi, gh, h_f, h_bf);

    // logits -> out[:, t, :]  (ldc spans the full [T-1, V1] row per batch)
    gemm_bf16<1><<<dim3(kV1 / 16, kB / 16), 32, 0, stream>>>(
        h_bf, lgt_bf, logit_b, out + (size_t)t * kV1, (__bf16*)nullptr,
        kV1, kH, kH, kH, (long long)(kT - 1) * kV1);

    log_softmax_kernel<<<kB, 256, 0, stream>>>(out, t);
  }
  (void)in_sizes; (void)n_in; (void)out_size; (void)ws_size;
}
